// ConservativeDynamicCurvatureMLP_50611894616640
// MI455X (gfx1250) — matrix operand order repaired and checked by reading, not compiled, not run on hardware
//
#include <hip/hip_runtime.h>
#include <math.h>

typedef __attribute__((ext_vector_type(16))) __bf16 v16bf;
typedef __attribute__((ext_vector_type(8)))  float  v8f;

#define B_DIM  8192
#define IN_DIM 3072
#define HID    4096
#define OUT_N  1000
#define OUT_P  1024
#define MIN_C  0.0005f
#define MAX_C  0.002f
#define T_FAC  0.7f
#define EPS_F  1e-7f

// ---------------------------------------------------------------- conversions
__global__ __launch_bounds__(256) void f32_to_bf16_kernel(const float* __restrict__ s,
                                                          __bf16* __restrict__ d, size_t n) {
  size_t i = (size_t)blockIdx.x * blockDim.x + threadIdx.x;
  size_t st = (size_t)gridDim.x * blockDim.x;
  for (; i < n; i += st) d[i] = (__bf16)s[i];
}

__global__ __launch_bounds__(256) void wo_pad_kernel(const float* __restrict__ s,
                                                     __bf16* __restrict__ d) {
  size_t n = (size_t)HID * OUT_P;
  size_t i = (size_t)blockIdx.x * blockDim.x + threadIdx.x;
  size_t st = (size_t)gridDim.x * blockDim.x;
  for (; i < n; i += st) {
    int r = (int)(i >> 10), c = (int)(i & 1023);
    d[i] = (c < OUT_N) ? (__bf16)s[(size_t)r * OUT_N + c] : (__bf16)0.0f;
  }
}

// ---------------------------------------------------------------- curvature predictor
__global__ __launch_bounds__(256) void curvature_kernel(
    const float* __restrict__ x, const float* __restrict__ w1, const float* __restrict__ b1,
    const float* __restrict__ w2, const float* __restrict__ b2, float* __restrict__ c_sum) {
  const int row = blockIdx.x;
  const float* xr = x + (size_t)row * IN_DIM;
  __shared__ float xs[1024];
  __shared__ float cph[16];
  const int j   = threadIdx.x >> 4;
  const int l16 = threadIdx.x & 15;
  float acc = 0.0f;
  for (int k0 = 0; k0 < IN_DIM; k0 += 1024) {
    __syncthreads();
    for (int i = threadIdx.x; i < 1024; i += 256) xs[i] = xr[k0 + i];
    __syncthreads();
    const float* wrow = w1 + (size_t)j * IN_DIM + k0;
    for (int kk = l16; kk < 1024; kk += 16) acc = fmaf(xs[kk], wrow[kk], acc);
  }
  for (int off = 8; off; off >>= 1) acc += __shfl_down(acc, off, 16);
  if (l16 == 0) cph[j] = acc;
  __syncthreads();
  if (threadIdx.x == 0) {
    float s = b2[0];
    for (int jj = 0; jj < 16; ++jj) {
      float v = cph[jj] + b1[jj];
      v = v > 0.0f ? v : 0.0f;
      s = fmaf(v, w2[jj], s);
    }
    float cpred  = 1.0f / (1.0f + expf(-s));
    float crange = MIN_C + (MAX_C - MIN_C) * cpred;
    atomicAdd(c_sum, crange);
  }
}

__global__ void c_final_kernel(const float* __restrict__ c_sum, float* __restrict__ c_avg) {
  float v = (*c_sum) * (1.0f / (float)B_DIM);
  *c_avg = fminf(fmaxf(v, MIN_C), MAX_C);
}

// ---------------------------------------------------------------- WMMA GEMM
// XOR chunk swizzle: element k of row r lives in 16B chunk (k>>3)^(r&3)^((r>>2)&3).
// Keeps every fragment read a single aligned ds_load_b128, conflict-free across lanes.
__device__ __forceinline__ int swz(int row, int cb) {
  return cb ^ (row & 3) ^ ((row >> 2) & 3);
}

// EPI 0: tanh -> Cf (f32, ld=N) and Cb (bf16, ld=N)
// EPI 1: sigmoid -> Cf (f32, ld=N)
// EPI 2: bias only, guard n < Nvalid -> Cf (f32, ld=Nvalid)
template <int EPI>
__global__ __launch_bounds__(256) void gemm_bf16_kernel(
    const __bf16* __restrict__ A, const __bf16* __restrict__ Bm,
    const float* __restrict__ bias, float* __restrict__ Cf, __bf16* __restrict__ Cb,
    int M, int N, int K, int Nvalid) {
  __shared__ __bf16 As[2][128 * 32];   // [M][K] tiles, double-buffered
  __shared__ __bf16 Bs[2][128 * 32];   // [N][K] tiles (transposed), double-buffered
  const int tid   = threadIdx.x;
  const int lane  = tid & 31;
  const int wid   = tid >> 5;
  const int waveM = wid >> 2;          // 0..1 -> 64-row slab
  const int waveN = wid & 3;           // 0..3 -> 32-col slab
  const int lrow  = lane & 15;
  const int hsel  = lane >> 4;
  const int m0 = blockIdx.y * 128;
  const int n0 = blockIdx.x * 128;

  const unsigned asBase[2] = { (unsigned)(size_t)&As[0][0], (unsigned)(size_t)&As[1][0] };

  auto stage = [&](int kt, int buf) {
    const int kbase = kt << 5;
    // A tile: async global->LDS, 16B per lane, no VGPR staging (ASYNCcnt tracked)
#pragma unroll
    for (int r = 0; r < 2; ++r) {
      int idx = tid + (r << 8);            // 0..511 -> one 16B chunk each
      int row = idx >> 2, cb = idx & 3;
      unsigned lds  = asBase[buf] + (unsigned)((row * 32 + (swz(row, cb) << 3)) * 2);
      unsigned goff = (unsigned)(((size_t)(m0 + row) * K + kbase + (cb << 3)) * 2);
      asm volatile("global_load_async_to_lds_b128 %0, %1, %2"
                   :: "v"(lds), "v"(goff), "s"(A) : "memory");
    }
    // B tile: read 2 K-rows x 8 cols, pack k/k+1 pairs, store transposed as b32
    const int kp  = tid >> 4;              // k = 2*kp
    const int col = (tid & 15) << 3;
    const __bf16* src = Bm + (size_t)(kbase + 2 * kp) * N + n0 + col;
    uint4 va = *(const uint4*)src;
    uint4 vb = *(const uint4*)(src + N);
    const unsigned short* ha = (const unsigned short*)&va;
    const unsigned short* hb = (const unsigned short*)&vb;
    const int cb = kp >> 2;
    const int within = (2 * kp) & 7;
#pragma unroll
    for (int jj = 0; jj < 8; ++jj) {
      int n = col + jj;
      unsigned w = (unsigned)ha[jj] | ((unsigned)hb[jj] << 16);
      *(unsigned*)&Bs[buf][n * 32 + (swz(n, cb) << 3) + within] = w;
    }
  };

  v8f acc[4][2] = {};
  const int ksteps = K >> 5;

  stage(0, 0);
  asm volatile("s_wait_asynccnt 0x0" ::: "memory");
  __syncthreads();

  for (int kt = 0; kt < ksteps; ++kt) {
    const int buf = kt & 1;
    if (kt + 1 < ksteps) stage(kt + 1, buf ^ 1);   // overlap with WMMA below

    // A frag (ISA 7.12.2): lanes0-15 K=0..7|16..23, lanes16-31 K=8..15|24..31
    v16bf afr[4];
#pragma unroll
    for (int i = 0; i < 4; ++i) {
      const int m = waveM * 64 + i * 16 + lrow;
      union { uint4 q[2]; v16bf v; } f;
      f.q[0] = *(const uint4*)&As[buf][m * 32 + (swz(m, hsel) << 3)];
      f.q[1] = *(const uint4*)&As[buf][m * 32 + (swz(m, 2 + hsel) << 3)];
      afr[i] = f.v;
    }
    // B frag: lane n = lane%16; lanes0-15 K=0..7|16..23, lanes16-31 K=8..15|24..31
    v16bf bfr[2];
#pragma unroll
    for (int j = 0; j < 2; ++j) {
      const int n = waveN * 32 + j * 16 + lrow;
      union { uint4 q[2]; v16bf v; } f;
      f.q[0] = *(const uint4*)&Bs[buf][n * 32 + (swz(n, hsel) << 3)];
      f.q[1] = *(const uint4*)&Bs[buf][n * 32 + (swz(n, 2 + hsel) << 3)];
      bfr[j] = f.v;
    }
#pragma unroll
    for (int i = 0; i < 4; ++i)
#pragma unroll
      for (int j = 0; j < 2; ++j)
        acc[i][j] = __builtin_amdgcn_wmma_f32_16x16x32_bf16(
            false, afr[i], false, bfr[j], (short)0, acc[i][j], false, false);

    asm volatile("s_wait_asynccnt 0x0" ::: "memory");  // next tile landed in LDS
    __syncthreads();
  }

  // epilogue: C/D layout — lane n = lane%16, VGPR r -> row r + 8*hsel
#pragma unroll
  for (int i = 0; i < 4; ++i) {
#pragma unroll
    for (int j = 0; j < 2; ++j) {
      const int n  = n0 + waveN * 32 + j * 16 + lrow;
      const int mb = m0 + waveM * 64 + i * 16 + hsel * 8;
      const float bn = (EPI == 2) ? ((n < Nvalid) ? bias[n] : 0.0f) : bias[n];
#pragma unroll
      for (int r = 0; r < 8; ++r) {
        float v = acc[i][j][r] + bn;
        if (EPI == 0) {
          v = tanhf(v);
          size_t off = (size_t)(mb + r) * N + n;
          Cf[off] = v;
          Cb[off] = (__bf16)v;
        } else if (EPI == 1) {
          v = 1.0f / (1.0f + expf(-v));
          Cf[(size_t)(mb + r) * N + n] = v;
        } else {
          if (n < Nvalid) Cf[(size_t)(mb + r) * Nvalid + n] = v;
        }
      }
    }
  }
}

// ---------------------------------------------------------------- Poincare layer
// z_i = P*h_i + Q*u_i with row scalars from h2,u2,hu (closed-form Mobius collapse).
__global__ __launch_bounds__(256) void poincare_kernel(
    const float* __restrict__ h, const float* __restrict__ u,
    const float* __restrict__ cavg, __bf16* __restrict__ zb, int* __restrict__ nanflag) {
  const int row = blockIdx.x, tid = threadIdx.x;
  const float* hr = h + (size_t)row * HID;
  const float* ur = u + (size_t)row * HID;
  float h2 = 0.0f, u2 = 0.0f, hu = 0.0f;
  for (int i = tid; i < HID; i += 256) {
    float a = hr[i], b = ur[i];
    h2 = fmaf(a, a, h2); u2 = fmaf(b, b, u2); hu = fmaf(a, b, hu);
  }
  for (int off = 16; off; off >>= 1) {
    h2 += __shfl_down(h2, off); u2 += __shfl_down(u2, off); hu += __shfl_down(hu, off);
  }
  __shared__ float red[8][3];
  __shared__ float sPQ[2];
  const int wv = tid >> 5, lane = tid & 31;
  if (lane == 0) { red[wv][0] = h2; red[wv][1] = u2; red[wv][2] = hu; }
  __syncthreads();
  if (tid == 0) {
    h2 = 0.0f; u2 = 0.0f; hu = 0.0f;
    for (int w = 0; w < 8; ++w) { h2 += red[w][0]; u2 += red[w][1]; hu += red[w][2]; }
    const float c = *cavg, sc = sqrtf(c);
    float a = 1.0f - 2.0f * c * hu + c * u2;
    float b = 1.0f - c * h2;
    float d = 1.0f - 2.0f * c * hu + c * c * h2 * u2 + EPS_F;
    float w2 = (a * a * h2 - 2.0f * a * b * hu + b * b * u2) / (d * d);
    float norm = fmaxf(sqrtf(fmaxf(w2, 0.0f)), EPS_F);
    float arg  = fminf(fmaxf(sc * norm, 0.0f), 1.0f - 1e-5f);
    float ath  = 0.5f * logf((1.0f + arg) / (1.0f - arg));
    float s    = tanhf(T_FAC * ath) / (sc * norm);
    float p = -s * a / d, q = s * b / d;
    float m2 = s * s * w2;
    float hm = p * h2 + q * hu;
    float alpha = 1.0f + 2.0f * c * hm + c * m2;
    float beta  = 1.0f - c * h2;
    float den   = 1.0f + 2.0f * c * hm + c * c * h2 * m2 + EPS_F;
    float P = (alpha + beta * p) / den;
    float Q = beta * q / den;
    if (isnan(P) || isnan(Q)) atomicOr(nanflag, 1);
    sPQ[0] = P; sPQ[1] = Q;
  }
  __syncthreads();
  const float P = sPQ[0], Q = sPQ[1];
  __bf16* zr = zb + (size_t)row * HID;
  for (int i = tid; i < HID; i += 256) zr[i] = (__bf16)(fmaf(P, hr[i], Q * ur[i]));
}

__global__ __launch_bounds__(256) void fixup_kernel(const float* __restrict__ h,
                                                   __bf16* __restrict__ zb,
                                                   const int* __restrict__ flag, size_t n) {
  if (*flag == 0) return;
  size_t i = (size_t)blockIdx.x * blockDim.x + threadIdx.x;
  size_t st = (size_t)gridDim.x * blockDim.x;
  for (; i < n; i += st) zb[i] = (__bf16)h[i];
}

// ---------------------------------------------------------------- launch
extern "C" void kernel_launch(void* const* d_in, const int* in_sizes, int n_in,
                              void* d_out, int out_size, void* d_ws, size_t ws_size,
                              hipStream_t stream) {
  const float* x     = (const float*)d_in[0];
  const float* W1    = (const float*)d_in[1];
  const float* b1    = (const float*)d_in[2];
  const float* W2    = (const float*)d_in[3];
  const float* b2    = (const float*)d_in[4];
  const float* Wo    = (const float*)d_in[5];
  const float* bo    = (const float*)d_in[6];
  const float* cp_w1 = (const float*)d_in[7];
  const float* cp_b1 = (const float*)d_in[8];
  const float* cp_w2 = (const float*)d_in[9];
  const float* cp_b2 = (const float*)d_in[10];
  float* out = (float*)d_out;

  char* ws = (char*)d_ws;
  size_t off = 0;
  auto alloc = [&](size_t bytes) -> void* {
    off = (off + 255) & ~(size_t)255;
    void* p = ws + off;
    off += bytes;
    return p;
  };
  float* c_sum   = (float*)alloc(16);
  int*   nanflag = (int*)((char*)c_sum + 4);
  float* c_avg   = (float*)((char*)c_sum + 8);
  __bf16* xb  = (__bf16*)alloc((size_t)B_DIM * IN_DIM * 2);
  __bf16* W1b = (__bf16*)alloc((size_t)IN_DIM * HID * 2);
  __bf16* W2b = (__bf16*)alloc((size_t)HID * HID * 2);
  __bf16* Wob = (__bf16*)alloc((size_t)HID * OUT_P * 2);
  float*  hf  = (float*)alloc((size_t)B_DIM * HID * 4);
  __bf16* hb  = (__bf16*)alloc((size_t)B_DIM * HID * 2);
  float*  uf  = (float*)alloc((size_t)B_DIM * HID * 4);
  __bf16* zbb = (__bf16*)alloc((size_t)B_DIM * HID * 2);

  hipMemsetAsync(c_sum, 0, 16, stream);

  f32_to_bf16_kernel<<<2048, 256, 0, stream>>>(x,  xb,  (size_t)B_DIM * IN_DIM);
  f32_to_bf16_kernel<<<2048, 256, 0, stream>>>(W1, W1b, (size_t)IN_DIM * HID);
  f32_to_bf16_kernel<<<2048, 256, 0, stream>>>(W2, W2b, (size_t)HID * HID);
  wo_pad_kernel<<<2048, 256, 0, stream>>>(Wo, Wob);

  curvature_kernel<<<B_DIM, 256, 0, stream>>>(x, cp_w1, cp_b1, cp_w2, cp_b2, c_sum);
  c_final_kernel<<<1, 1, 0, stream>>>(c_sum, c_avg);

  gemm_bf16_kernel<0><<<dim3(HID / 128, B_DIM / 128), 256, 0, stream>>>(
      xb, W1b, b1, hf, hb, B_DIM, HID, IN_DIM, HID);
  gemm_bf16_kernel<1><<<dim3(HID / 128, B_DIM / 128), 256, 0, stream>>>(
      hb, W2b, b2, uf, nullptr, B_DIM, HID, HID, HID);

  poincare_kernel<<<B_DIM, 256, 0, stream>>>(hf, uf, c_avg, zbb, nanflag);
  fixup_kernel<<<4096, 256, 0, stream>>>(hf, zbb, nanflag, (size_t)B_DIM * HID);

  gemm_bf16_kernel<2><<<dim3(OUT_P / 128, B_DIM / 128), 256, 0, stream>>>(
      zbb, Wob, bo, out, nullptr, B_DIM, OUT_P, HID, OUT_N);
}
